// RecurrentEncoder_12232066859256
// MI455X (gfx1250) — matrix operand order repaired and checked by reading, not compiled, not run on hardware
//
#include <hip/hip_runtime.h>

typedef _Float16 v16h __attribute__((ext_vector_type(16)));
typedef _Float16 v8h  __attribute__((ext_vector_type(8)));
typedef float    v8f  __attribute__((ext_vector_type(8)));
typedef int      v4i_vs __attribute__((vector_size(4 * sizeof(int))));

#define VOCABN 32000
#define DD  512
#define HH  512
#define LL  2
#define NDIR 2
#define BB  64
#define SS  512
#define PADTOK 0

// ---------------- embedding gather -> fp16, time-major (S,B,D) ----------------
__global__ void embed_gather(const int* __restrict__ tokens,
                             const float* __restrict__ table,
                             _Float16* __restrict__ e) {
  int idx = blockIdx.x;                 // b*S + t  (matches tokens flat layout)
  int b = idx / SS, t = idx % SS;
  int tok = tokens[idx];
  const float* src = table + (size_t)tok * DD;
  _Float16* dst = e + ((size_t)t * BB + b) * DD;
  for (int d = threadIdx.x; d < DD; d += blockDim.x) {
    float v = (tok == PADTOK) ? 0.0f : src[d];   // padding_idx row forced to 0
    dst[d] = (_Float16)v;
  }
}

// ---------------- fp32 -> fp16 weight convert ----------------
__global__ void w_to_f16(const float* __restrict__ src, _Float16* __restrict__ dst, int n) {
  int i = blockIdx.x * blockDim.x + threadIdx.x;
  if (i < n) dst[i] = (_Float16)src[i];
}

// ---------------- WMMA fragment helpers (wave32 layouts per CDNA5 ISA) ----------------
// A-matrix 16x32 f16: lanes 0-15 row M=lane hold K{0..7,16..23}; lanes 16-31 hold K{8..15,24..31}
__device__ __forceinline__ v16h loadA(const _Float16* base /* tile origin, ld=HH */, int lane) {
  int row = lane & 15;
  int kb  = (lane & 16) ? 8 : 0;
  const _Float16* p = base + (size_t)row * HH + kb;
  v8h lo = *(const v8h*)(p);
  v8h hi = *(const v8h*)(p + 16);
  v16h r;
#pragma unroll
  for (int i = 0; i < 8; ++i) { r[i] = lo[i]; r[i + 8] = hi[i]; }
  return r;
}

// B-matrix 32x16 f16 for x @ W^T: b[k][n] = W[n][k], lane col = lane&15,
// lanes 0-15 hold K{0..7,16..23}, lanes 16-31 hold K{8..15,24..31} (matches A)
__device__ __forceinline__ v16h loadB(const _Float16* __restrict__ W, int ntile, int kc, int lane) {
  int col = lane & 15;
  int kb  = (lane & 16) ? 8 : 0;
  const _Float16* p = W + (size_t)(ntile * 16 + col) * HH + kc * 32 + kb;
  v8h lo = *(const v8h*)(p);
  v8h hi = *(const v8h*)(p + 16);
  v16h r;
#pragma unroll
  for (int i = 0; i < 8; ++i) { r[i] = lo[i]; r[i + 8] = hi[i]; }
  return r;
}

#define WMMA_F16(a, b, c) \
  __builtin_amdgcn_wmma_f32_16x16x32_f16(false, (a), false, (b), (short)0, (c), false, false)

__device__ __forceinline__ float sigm(float x)  { return 1.0f / (1.0f + __expf(-x)); }
__device__ __forceinline__ float tanh_f(float x){ return 1.0f - 2.0f / (__expf(2.0f * x) + 1.0f); }

#if defined(__HIP_DEVICE_COMPILE__) && \
    __has_builtin(__builtin_amdgcn_global_load_async_to_lds_b128)
#define HAVE_ASYNC_LDS 1
#else
#define HAVE_ASYNC_LDS 0
#endif

// ---------------- persistent recurrent kernel: one block per direction ----------------
__global__ __launch_bounds__(1024) void gru_recurrent(
    const _Float16* __restrict__ e,       // (S,B,D) fp16
    const int* __restrict__ lengths,      // (B)
    const _Float16* __restrict__ Wall,    // 6 arrays of (NDIR,L,H,H) fp16
    const float* __restrict__ bxz,        // (NDIR,L,H)
    const float* __restrict__ bxr,
    const float* __restrict__ bxh,
    float* __restrict__ out)              // (B,S,2H) fp32
{
  extern __shared__ _Float16 smem[];
  _Float16* H0   = smem;                         // 64x512 layer-0 hidden (fp16)
  _Float16* H1   = smem + (size_t)BB * HH;       // 64x512 layer-1 hidden
  _Float16* TMPX = smem + (size_t)2 * BB * HH;   // x stage (layer-0 input)
  _Float16* TMPR = smem + (size_t)3 * BB * HH;   // r*h stage

  const int dir  = blockIdx.x;
  const int tid  = threadIdx.x;
  const int lane = tid & 31;
  const int nt   = tid >> 5;                     // 32 waves -> 32 N-tiles (512 cols)
  const int col  = lane & 15;
  const int rb   = (lane & 16) ? 8 : 0;          // C-fragment row base

  // zero hidden state
  for (int i = tid; i < 2 * BB * HH; i += blockDim.x) smem[i] = (_Float16)0;
  __syncthreads();

  const size_t WSZ = (size_t)NDIR * LL * HH * HH;
  const size_t MO  = (size_t)HH * HH;
  const int n = nt * 16 + col;                   // output feature owned by this lane

#pragma unroll 1
  for (int s = 0; s < SS; ++s) {
    const int tsrc = dir ? (SS - 1 - s) : s;     // bwd reads reversed time; t_out == tsrc

    // ---- stage x for layer 0 (64x512 fp16 = 64KB contiguous) ----
#if HAVE_ASYNC_LDS
    {
      const char* gsrc = (const char*)(e + (size_t)tsrc * BB * DD);
      char*       ldst = (char*)TMPX;
      for (int i = tid; i < (BB * DD) / 8; i += blockDim.x) {
        __builtin_amdgcn_global_load_async_to_lds_b128(
            (__attribute__((address_space(1))) v4i_vs*)(gsrc + 16 * (size_t)i),
            (__attribute__((address_space(3))) v4i_vs*)(ldst + 16 * (size_t)i),
            0, 0);
      }
#if __has_builtin(__builtin_amdgcn_s_wait_asynccnt)
      __builtin_amdgcn_s_wait_asynccnt(0);
#else
      asm volatile("s_wait_asynccnt 0" ::: "memory");
#endif
    }
#else
    {
      const v8h* src = (const v8h*)(e + (size_t)tsrc * BB * DD);
      v8h* dst = (v8h*)TMPX;
      for (int i = tid; i < (BB * DD) / 8; i += blockDim.x) dst[i] = src[i];
    }
#endif
    if (s + 1 < SS) {                            // prefetch next step's x into cache
      const int tn = dir ? (SS - 2 - s) : (s + 1);
      __builtin_prefetch(e + (size_t)tn * BB * DD + (size_t)tid * 16, 0, 1);
    }
    __syncthreads();

    const _Float16* xls = TMPX;
    _Float16*       hls = H0;

#pragma unroll 1
    for (int l = 0; l < LL; ++l) {
      const size_t dlo = (size_t)(dir * LL + l) * MO;
      const _Float16* Wxz_ = Wall + 0 * WSZ + dlo;
      const _Float16* Whz_ = Wall + 1 * WSZ + dlo;
      const _Float16* Wxr_ = Wall + 2 * WSZ + dlo;
      const _Float16* Whr_ = Wall + 3 * WSZ + dlo;
      const _Float16* Wxh_ = Wall + 4 * WSZ + dlo;
      const _Float16* Whh_ = Wall + 5 * WSZ + dlo;
      const int bidx = (dir * LL + l) * HH + n;

      // ---- fused z/r gate GEMMs: share A fragments across 4 weight streams ----
      v8f gz[4] = {};
      v8f gr[4] = {};
#pragma unroll 2
      for (int kc = 0; kc < 16; ++kc) {
        v16h ax[4], ah[4];
#pragma unroll
        for (int m = 0; m < 4; ++m)
          ax[m] = loadA(xls + (size_t)(m * 16) * HH + kc * 32, lane);
#pragma unroll
        for (int m = 0; m < 4; ++m)
          ah[m] = loadA(hls + (size_t)(m * 16) * HH + kc * 32, lane);

        v16h b;
        b = loadB(Wxz_, nt, kc, lane);
#pragma unroll
        for (int m = 0; m < 4; ++m) gz[m] = WMMA_F16(ax[m], b, gz[m]);
        b = loadB(Whz_, nt, kc, lane);
#pragma unroll
        for (int m = 0; m < 4; ++m) gz[m] = WMMA_F16(ah[m], b, gz[m]);
        b = loadB(Wxr_, nt, kc, lane);
#pragma unroll
        for (int m = 0; m < 4; ++m) gr[m] = WMMA_F16(ax[m], b, gr[m]);
        b = loadB(Whr_, nt, kc, lane);
#pragma unroll
        for (int m = 0; m < 4; ++m) gr[m] = WMMA_F16(ah[m], b, gr[m]);
      }

      // r = sigmoid(gr + bxr); stage r*h (fp16) for the Whh GEMM, in-fragment
      const float brv = bxr[bidx];
#pragma unroll
      for (int m = 0; m < 4; ++m) {
#pragma unroll
        for (int v = 0; v < 8; ++v) {
          int bi = m * 16 + rb + v;              // batch row
          float r  = sigm(gr[m][v] + brv);
          float hv = (float)hls[(size_t)bi * HH + n];
          TMPR[(size_t)bi * HH + n] = (_Float16)(r * hv);
        }
      }
      __syncthreads();                           // rh visible to all waves

      // ---- fused h~ GEMM: x@Wxh^T + (r*h)@Whh^T ----
      v8f gh[4] = {};
#pragma unroll 2
      for (int kc = 0; kc < 16; ++kc) {
        v16h ax[4], ar[4];
#pragma unroll
        for (int m = 0; m < 4; ++m)
          ax[m] = loadA(xls + (size_t)(m * 16) * HH + kc * 32, lane);
#pragma unroll
        for (int m = 0; m < 4; ++m)
          ar[m] = loadA(TMPR + (size_t)(m * 16) * HH + kc * 32, lane);

        v16h b;
        b = loadB(Wxh_, nt, kc, lane);
#pragma unroll
        for (int m = 0; m < 4; ++m) gh[m] = WMMA_F16(ax[m], b, gh[m]);
        b = loadB(Whh_, nt, kc, lane);
#pragma unroll
        for (int m = 0; m < 4; ++m) gh[m] = WMMA_F16(ar[m], b, gh[m]);
      }

      const float bzv = bxz[bidx];
      const float bhv = bxh[bidx];
#pragma unroll
      for (int m = 0; m < 4; ++m) {
#pragma unroll
        for (int v = 0; v < 8; ++v) {
          int bi = m * 16 + rb + v;
          bool active = s < lengths[bi];         // step-index mask (same both dirs)
          float hold = (float)hls[(size_t)bi * HH + n];
          float z    = sigm(gz[m][v] + bzv);
          float ht   = tanh_f(gh[m][v] + bhv);
          float hnew = active ? ((1.0f - z) * hold + z * ht) : hold;
          hls[(size_t)bi * HH + n] = (_Float16)hnew;
          if (l == LL - 1)
            out[((size_t)bi * SS + tsrc) * (2 * HH) + dir * HH + n] = active ? hnew : 0.0f;
        }
      }
      __syncthreads();                           // h_new visible before next layer/step

      xls = H0;                                  // layer-1 input = layer-0 new hidden
      hls = H1;
    }
  }
}

extern "C" void kernel_launch(void* const* d_in, const int* in_sizes, int n_in,
                              void* d_out, int out_size, void* d_ws, size_t ws_size,
                              hipStream_t stream) {
  const int*   tokens  = (const int*)d_in[0];
  const int*   lengths = (const int*)d_in[1];
  const float* table   = (const float*)d_in[2];
  const float* Wxz = (const float*)d_in[3];
  const float* bxz = (const float*)d_in[4];
  const float* Whz = (const float*)d_in[5];
  const float* Wxr = (const float*)d_in[6];
  const float* bxr = (const float*)d_in[7];
  const float* Whr = (const float*)d_in[8];
  const float* Wxh = (const float*)d_in[9];
  const float* bxh = (const float*)d_in[10];
  const float* Whh = (const float*)d_in[11];

  _Float16* e  = (_Float16*)d_ws;                       // S*B*D fp16  (32 MB)
  _Float16* Wf = e + (size_t)SS * BB * DD;              // 6 * NDIR*L*H*H fp16 (12 MB)

  const int WN = NDIR * LL * HH * HH;
  const float* srcs[6] = {Wxz, Whz, Wxr, Whr, Wxh, Whh};
  for (int i = 0; i < 6; ++i)
    w_to_f16<<<(WN + 255) / 256, 256, 0, stream>>>(srcs[i], Wf + (size_t)i * WN, WN);

  embed_gather<<<BB * SS, 256, 0, stream>>>(tokens, table, e);

  const int shmem = 4 * BB * HH * (int)sizeof(_Float16);    // 256 KB of the 320 KB WGP LDS
  (void)hipFuncSetAttribute((const void*)gru_recurrent,
                            hipFuncAttributeMaxDynamicSharedMemorySize, shmem);
  gru_recurrent<<<NDIR, 1024, shmem, stream>>>(e, lengths, Wf, bxz, bxr, bxh, (float*)d_out);
}
